// ConditionalRationalQuadraticTransformation_38319698215023
// MI455X (gfx1250) — matrix operand order repaired and checked by reading, not compiled, not run on hardware
//
#include <hip/hip_runtime.h>

typedef __attribute__((ext_vector_type(16))) _Float16 v16h;
typedef __attribute__((ext_vector_type(8)))  _Float16 v8h;
typedef __attribute__((ext_vector_type(4)))  _Float16 v4h;
typedef __attribute__((ext_vector_type(8)))  float    v8f;
typedef __attribute__((ext_vector_type(4)))  float    v4f;

#define M_TILE     128
#define ACT_STRIDE 264      // 256 + 8 halves pad -> 528B rows, 16B aligned, bank-spread
#define WST_STRIDE 136      // 128 + 8 halves pad -> 272B rows (68 dwords % 64 = 4)
#define OUT_STRIDE 193      // fp32 logits row stride (odd dword stride -> no bank conflicts)
#define TAILB      5.0f
#define MIN_BIN    0.001f
#define MIN_DERIV  0.001f
#define SM_SCALE   0.936f   // 1 - MIN_BIN * 64

// workspace layout (in halves)
#define W0T_OFF 0                       // [256][128] f16  (out-neuron major)
#define W1T_OFF 32768                   // [256][256]
#define W2T_OFF (32768 + 65536)         // [256][256]
#define W3T_OFF (32768 + 2*65536)       // [192][256], rows >=191 zero
#define WS_HALVES (32768 + 2*65536 + 49152)

// ---------------------------------------------------------------------------
// Weight prep: fp32 -> f16, transposed (row = output neuron, contiguous K)
// ---------------------------------------------------------------------------
__global__ void prep_weights_f16t(const float* __restrict__ W0, const float* __restrict__ W1,
                                  const float* __restrict__ W2, const float* __restrict__ W3,
                                  _Float16* __restrict__ ws) {
  int i = blockIdx.x * blockDim.x + threadIdx.x;
  if (i < 32768) {                                   // W0T: 256 x 128
    int n = i >> 7, k = i & 127;
    ws[W0T_OFF + i] = (_Float16)W0[k * 256 + n];
  } else if (i < 32768 + 65536) {                    // W1T: 256 x 256
    int j = i - 32768; int n = j >> 8, k = j & 255;
    ws[W1T_OFF + j] = (_Float16)W1[k * 256 + n];
  } else if (i < 32768 + 2 * 65536) {                // W2T: 256 x 256
    int j = i - (32768 + 65536); int n = j >> 8, k = j & 255;
    ws[W2T_OFF + j] = (_Float16)W2[k * 256 + n];
  } else if (i < WS_HALVES) {                        // W3T: 192 x 256 (pad 191->192)
    int j = i - (32768 + 2 * 65536); int n = j >> 8, k = j & 255;
    ws[W3T_OFF + j] = (n < 191) ? (_Float16)W3[k * 191 + n] : (_Float16)0.0f;
  }
}

// ---------------------------------------------------------------------------
// CDNA5 async global -> LDS copy (ASYNCcnt), 16B per lane.
// LDS aperture uses only addr[63:32]; addr[31:0] is the wave-relative LDS
// address, so truncating a flat shared pointer yields the VDST operand.
// ---------------------------------------------------------------------------
__device__ __forceinline__ void async_g2l_b128(void* ldsDst, const void* gSrc) {
  unsigned off = (unsigned)(unsigned long long)ldsDst;
  asm volatile("global_load_async_to_lds_b128 %0, %1, off"
               :: "v"(off), "v"(gSrc) : "memory");
}
__device__ __forceinline__ void wait_async0() {
  asm volatile("s_wait_asynccnt 0x0" ::: "memory");
}

__device__ __forceinline__ v16h cat8(v8h lo, v8h hi) {
  return __builtin_shufflevector(lo, hi, 0,1,2,3,4,5,6,7,8,9,10,11,12,13,14,15);
}

// Stage W^T[0:rows, k0:k0+128) into LDS (stride WST_STRIDE), async.
__device__ __forceinline__ void stage_w(const _Float16* __restrict__ wT, int rows, int kin,
                                        int k0, _Float16* __restrict__ wst, int tid) {
  for (int i = tid; i < rows * 16; i += 512) {       // 16 x 16B chunks per row
    int n = i >> 4, c = i & 15;
    async_g2l_b128(wst + n * WST_STRIDE + c * 8, wT + (size_t)n * kin + k0 + c * 8);
  }
  wait_async0();
}

// ---------------------------------------------------------------------------
// One wave: 16 x (NT*16) tile over a K=128 slice. A from act LDS, B from
// staged-weight LDS.  A layout: lane m rows, halves [k+hi*8..+7],[+16..+23].
// B layout: lane n cols, halves [k+hi*8..+7],[+16..+23].
// ---------------------------------------------------------------------------
template<int NT>
__device__ __forceinline__ void gemm_slice(const _Float16* __restrict__ actIn, int k0,
                                           int rowBaseW, const _Float16* __restrict__ wst,
                                           int nBase, int lane, v8f (&acc)[NT]) {
  const int rlo = lane & 15;
  const int hi  = lane >> 4;
#pragma unroll
  for (int k = 0; k < 128; k += 32) {
    const _Float16* ap = actIn + (rowBaseW + rlo) * ACT_STRIDE + k0 + k + hi * 8;
    v16h a = cat8(*(const v8h*)(ap), *(const v8h*)(ap + 16));
#pragma unroll
    for (int n = 0; n < NT; ++n) {
      const _Float16* bp = wst + (nBase + n * 16 + rlo) * WST_STRIDE + k + hi * 8;
      v16h b = cat8(*(const v8h*)(bp), *(const v8h*)(bp + 16));
      acc[n] = __builtin_amdgcn_wmma_f32_16x16x32_f16(
          false, a, false, b, (short)0, acc[n], false, false);
    }
  }
}

// C/D layout: c[v], lane l: m = v + (l<16?0:8), n = l&15
template<int NT>
__device__ __forceinline__ void epi_relu_f16(const v8f (&acc)[NT], const float* __restrict__ bias,
                                             int nBase, _Float16* __restrict__ ldsOut,
                                             int rowBaseW, int lane) {
  const int col = lane & 15;
  const int hi  = lane >> 4;
#pragma unroll
  for (int n = 0; n < NT; ++n) {
    float bv = bias[nBase + n * 16 + col];
#pragma unroll
    for (int v = 0; v < 8; ++v) {
      float x = fmaxf(acc[n][v] + bv, 0.0f);
      int m = rowBaseW + v + hi * 8;
      ldsOut[m * ACT_STRIDE + nBase + n * 16 + col] = (_Float16)x;
    }
  }
}

__device__ __forceinline__ float softplusf(float x) {
  return fmaxf(x, 0.0f) + log1pf(__expf(-fabsf(x)));
}

// ---------------------------------------------------------------------------
// Fused: async-stage context -> 4-layer MLP (WMMA f16, LDS-staged weights,
// in-place activations) -> RQ-spline inverse per row.
// 512 threads = 16 waves: wave = (rowGroup<<1) | columnHalf
// ---------------------------------------------------------------------------
__global__ __launch_bounds__(512) void fused_mlp_spline(
    const float* __restrict__ Y, const float* __restrict__ ctx,
    const float* __restrict__ b0, const float* __restrict__ b1,
    const float* __restrict__ b2, const float* __restrict__ b3,
    const _Float16* __restrict__ ws, float* __restrict__ out, int Ntot) {
  __shared__ _Float16 smem[M_TILE * ACT_STRIDE + 256 * WST_STRIDE];  // 137216 B
  _Float16* actbuf = smem;
  _Float16* wstage = smem + M_TILE * ACT_STRIDE;

  const _Float16* w0t = ws + W0T_OFF;
  const _Float16* w1t = ws + W1T_OFF;
  const _Float16* w2t = ws + W2T_OFF;
  const _Float16* w3t = ws + W3T_OFF;

  const int tid  = threadIdx.x;
  const int lane = tid & 31;
  const int wave = tid >> 5;
  const int rowBlk   = blockIdx.x * M_TILE;
  const int rowBaseW = (wave >> 1) * 16;   // 0..112
  const int half     = wave & 1;           // column half

  // ---- async-stage raw fp32 context tile into wstage region (64KB) ----
  float* ctxStage = (float*)wstage;
  for (int i = tid; i < (M_TILE * 128) / 4; i += 512) {   // 16B chunks
    int gr = rowBlk + (i >> 5);
    if (gr < Ntot)
      async_g2l_b128(ctxStage + i * 4, ctx + (size_t)rowBlk * 128 + (size_t)i * 4);
  }
  wait_async0();
  __syncthreads();

  // ---- convert fp32 -> f16 into actbuf (stride ACT_STRIDE) ----
  for (int i = tid; i < (M_TILE * 128) / 4; i += 512) {
    int r  = i >> 5;
    int c4 = (i & 31) << 2;
    v4h hv; hv[0] = hv[1] = hv[2] = hv[3] = (_Float16)0.0f;
    if (rowBlk + r < Ntot) {
      v4f cv = *(const v4f*)(ctxStage + r * 128 + c4);
      hv[0] = (_Float16)cv[0]; hv[1] = (_Float16)cv[1];
      hv[2] = (_Float16)cv[2]; hv[3] = (_Float16)cv[3];
    }
    *(v4h*)(actbuf + r * ACT_STRIDE + c4) = hv;
  }
  __syncthreads();

  const v8f vzero = {0.f, 0.f, 0.f, 0.f, 0.f, 0.f, 0.f, 0.f};

  // ---- layer 0: 128 -> 256, relu, in-place ----
  {
    v8f acc[8]; for (int n = 0; n < 8; ++n) acc[n] = vzero;
    stage_w(w0t, 256, 128, 0, wstage, tid);
    __syncthreads();
    gemm_slice<8>(actbuf, 0, rowBaseW, wstage, half * 128, lane, acc);
    __syncthreads();                    // all reads of actbuf done
    epi_relu_f16<8>(acc, b0, half * 128, actbuf, rowBaseW, lane);
    __syncthreads();
  }

  // ---- layers 1 & 2: 256 -> 256, relu, in-place, two K-slices ----
  const _Float16* wmid[2]  = {w1t, w2t};
  const float*    bmid[2]  = {b1, b2};
  for (int L = 0; L < 2; ++L) {
    v8f acc[8]; for (int n = 0; n < 8; ++n) acc[n] = vzero;
    stage_w(wmid[L], 256, 256, 0, wstage, tid);
    __syncthreads();
    gemm_slice<8>(actbuf, 0, rowBaseW, wstage, half * 128, lane, acc);
    __syncthreads();                    // wstage free for restage
    stage_w(wmid[L], 256, 256, 128, wstage, tid);
    __syncthreads();
    gemm_slice<8>(actbuf, 128, rowBaseW, wstage, half * 128, lane, acc);
    __syncthreads();                    // all reads of actbuf & wstage done
    epi_relu_f16<8>(acc, bmid[L], half * 128, actbuf, rowBaseW, lane);
    __syncthreads();
  }

  // ---- layer 3: 256 -> 192(191), fp32 logits into smem base ----
  {
    v8f acc[6]; for (int n = 0; n < 6; ++n) acc[n] = vzero;
    stage_w(w3t, 192, 256, 0, wstage, tid);
    __syncthreads();
    gemm_slice<6>(actbuf, 0, rowBaseW, wstage, half * 96, lane, acc);
    __syncthreads();
    stage_w(w3t, 192, 256, 128, wstage, tid);
    __syncthreads();
    gemm_slice<6>(actbuf, 128, rowBaseW, wstage, half * 96, lane, acc);
    __syncthreads();                    // everyone done reading LDS
    float* o = (float*)smem;
    const int col = lane & 15;
    const int hi  = lane >> 4;
#pragma unroll
    for (int n = 0; n < 6; ++n) {
      int gc = half * 96 + n * 16 + col;
      float bv = (gc < 191) ? b3[gc] : 0.0f;
#pragma unroll
      for (int v = 0; v < 8; ++v) {
        int m = rowBaseW + v + hi * 8;
        o[m * OUT_STRIDE + gc] = acc[n][v] + bv;
      }
    }
  }
  __syncthreads();

  // ---- spline inverse: one thread per row ----
  if (tid < M_TILE) {
    int gRow = rowBlk + tid;
    if (gRow < Ntot) {
      const float* o  = (const float*)smem + tid * OUT_STRIDE;
      const float* uw = o;
      const float* uh = o + 64;
      const float* ud = o + 128;   // 63 valid

      float x  = Y[gRow];
      float xc = fminf(fmaxf(x, -TAILB), TAILB);

      // heights softmax + streaming cumsum -> bin index, ch[idx], heights[idx]
      float hmax = -1e30f;
      for (int i = 0; i < 64; ++i) hmax = fmaxf(hmax, uh[i]);
      float hsum = 0.0f;
      for (int i = 0; i < 64; ++i) hsum += __expf(uh[i] - hmax);
      float hinv = SM_SCALE / hsum;

      int   idx   = 63;
      float in_ch = -TAILB, in_h = 2.0f * TAILB;
      bool  found = false;
      float cum = 0.0f, chp = -TAILB;
      for (int i = 0; i < 64; ++i) {
        cum += MIN_BIN + __expf(uh[i] - hmax) * hinv;
        float chi = (i == 63) ? TAILB : (2.0f * TAILB * cum - TAILB);
        float loc = (i == 63) ? (TAILB + 1e-6f) : chi;
        if (!found) {
          if (xc >= loc) {
            chp = chi;
          } else {
            idx = i; in_ch = chp; in_h = chi - chp; found = true;
          }
        }
      }

      // widths softmax + cumsum up to idx -> cw[idx], widths[idx]
      float wmax = -1e30f;
      for (int i = 0; i < 64; ++i) wmax = fmaxf(wmax, uw[i]);
      float wsum = 0.0f;
      for (int i = 0; i < 64; ++i) wsum += __expf(uw[i] - wmax);
      float winv = SM_SCALE / wsum;

      float cumw = 0.0f, in_cw = -TAILB, cwn = TAILB;
      for (int i = 0; i <= idx; ++i) {
        float w = MIN_BIN + __expf(uw[i] - wmax) * winv;
        if (i == idx) in_cw = (i == 0) ? -TAILB : (2.0f * TAILB * cumw - TAILB);
        cumw += w;
        if (i == idx) cwn = (i == 63) ? TAILB : (2.0f * TAILB * cumw - TAILB);
      }
      float in_w = cwn - in_cw;

      // derivatives (padded ends = 1.0 exactly)
      float d0 = (idx == 0)  ? 1.0f : (MIN_DERIV + softplusf(ud[idx - 1]));
      float d1 = (idx == 63) ? 1.0f : (MIN_DERIV + softplusf(ud[idx]));

      float in_delta = in_h / in_w;
      float dsum = d0 + d1 - 2.0f * in_delta;
      float dy   = xc - in_ch;
      float qa   = dy * dsum + in_h * (in_delta - d0);
      float qb   = in_h * d0 - dy * dsum;
      float qc   = -in_delta * dy;
      float disc = qb * qb - 4.0f * qa * qc;
      float root = 2.0f * qc / (-qb - sqrtf(disc));
      float yout = root * in_w + in_cw;
      float tt   = root * (1.0f - root);
      float den  = in_delta + dsum * tt;
      float omr  = 1.0f - root;
      float dnum = in_delta * in_delta * (d1 * root * root + 2.0f * in_delta * tt + d0 * omr * omr);
      float lad  = __logf(dnum) - 2.0f * __logf(den);

      bool inside = (x >= -TAILB) && (x <= TAILB);
      out[gRow]        = inside ? yout : x;
      out[Ntot + gRow] = inside ? -lad : 0.0f;
    }
  }
}

// ---------------------------------------------------------------------------
extern "C" void kernel_launch(void* const* d_in, const int* in_sizes, int n_in,
                              void* d_out, int out_size, void* d_ws, size_t ws_size,
                              hipStream_t stream) {
  const float* Y   = (const float*)d_in[0];
  const float* ctx = (const float*)d_in[1];
  const float* W0  = (const float*)d_in[2];
  const float* b0  = (const float*)d_in[3];
  const float* W1  = (const float*)d_in[4];
  const float* b1  = (const float*)d_in[5];
  const float* W2  = (const float*)d_in[6];
  const float* b2  = (const float*)d_in[7];
  const float* W3  = (const float*)d_in[8];
  const float* b3  = (const float*)d_in[9];
  float*    out = (float*)d_out;
  _Float16* ws  = (_Float16*)d_ws;
  const int N = in_sizes[0];   // Y is (N,1)

  prep_weights_f16t<<<(WS_HALVES + 255) / 256, 256, 0, stream>>>(W0, W1, W2, W3, ws);

  const int blocks = (N + M_TILE - 1) / M_TILE;
  fused_mlp_spline<<<blocks, 512, 0, stream>>>(Y, ctx, b0, b1, b2, b3, ws, out, N);
}
